// STMAE_42502996361749
// MI455X (gfx1250) — matrix operand order repaired and checked by reading, not compiled, not run on hardware
//
#include <hip/hip_runtime.h>

// STMAE for MI455X (gfx1250, wave32, WMMA bf16 16x16x32)
// B=32 T=128 D=512 V=128 S=64 N=4 R=64 O=64

#define B_ 32
#define T_ 128
#define D_ 512
#define V_ 128
#define S_ 64
#define N_ 4
#define R_ 64
#define O_ 64
#define M_ (B_ * T_)   // 4096 (t*B + b rows)

#define INV_SQRT_S 0.125f

typedef __attribute__((ext_vector_type(16))) __bf16 bf16x16;
typedef __attribute__((ext_vector_type(8)))  float  floatx8;

__device__ __forceinline__ floatx8 zero8() {
  floatx8 z = {0.f, 0.f, 0.f, 0.f, 0.f, 0.f, 0.f, 0.f};
  return z;
}

__device__ __forceinline__ floatx8 wmma_bf16(bf16x16 a, bf16x16 b, floatx8 c) {
  return __builtin_amdgcn_wmma_f32_16x16x32_bf16(false, a, false, b, (short)0, c,
                                                 false, false);
}

// ---- WMMA fragment helpers (CDNA5 ISA 7.12.2 layouts, wave32) ----
// A (16x32 bf16): lane -> row M=lane&15, kb=(lane>>4)*8; K = kb..kb+7 and 16+kb..16+kb+7
__device__ __forceinline__ bf16x16 frag_a_rowptr(const float* rowp, int k0) {
  const int lane = threadIdx.x & 31;
  const int kb = (lane >> 4) * 8;
  bf16x16 a;
#pragma unroll
  for (int e = 0; e < 8; ++e) {
    a[e]     = (__bf16)rowp[k0 + kb + e];
    a[e + 8] = (__bf16)rowp[k0 + 16 + kb + e];
  }
  return a;
}

// B (32x16 bf16) from row-major fp32 W[K][Nc]: lane -> col N=lane&15, kb=(lane>>4)*8; K = kb..kb+7 and 16+kb..16+kb+7
__device__ __forceinline__ bf16x16 frag_b_global(const float* __restrict__ W, int ldb,
                                                 int k0, int c0) {
  const int lane = threadIdx.x & 31;
  const int n = lane & 15;
  const int kb = (lane >> 4) * 8;
  const float* p = W + (k0 + kb) * ldb + c0 + n;
  bf16x16 b;
#pragma unroll
  for (int e = 0; e < 16; ++e) b[e] = (__bf16)p[(e < 8 ? e : e + 8) * ldb];
  return b;
}

// C/D tile store: N = lane&15, M = i + (lane>>4)*8
template <int ACT, int OUT_MODE>
__device__ __forceinline__ void store_tile(float* __restrict__ C, int Nc, int r0, int c0,
                                           floatx8 acc, const float* __restrict__ bias) {
  const int lane = threadIdx.x & 31;
  const int nl = lane & 15, mb = (lane >> 4) * 8;
#pragma unroll
  for (int i = 0; i < 8; ++i) {
    int row = r0 + mb + i, col = c0 + nl;
    float x = acc[i] + bias[col];
    if (ACT) x = fmaxf(x, 0.f);
    if (OUT_MODE == 0) {
      C[row * Nc + col] = x;
    } else {                                       // row = t*B + b -> (B, V, T)
      int bb = row & (B_ - 1), tt = row >> 5;
      C[bb * (Nc * T_) + col * T_ + tt] = x;
    }
  }
}

// ---------------- generic batched GEMM: C = act(A @ W + bias) ----------------
// Each wave computes a 16x64 output strip: one A fragment feeds 4 WMMAs per k-step.
template <int ACT, int OUT_MODE>
__global__ void __launch_bounds__(128)
gemm_bf16_t(const float* __restrict__ A, const float* __restrict__ W,
            const float* __restrict__ bias, float* __restrict__ C,
            int M, int Nc, int K) {
  const int grpN = Nc >> 6;
  const int totGrp = (M >> 4) * grpN;
  const int grp = blockIdx.x * 4 + (threadIdx.x >> 5);
  if (grp >= totGrp) return;
  const int r0 = (grp / grpN) << 4;
  const int c0 = (grp % grpN) << 6;
  const int lane = threadIdx.x & 31;
  const float* rowp = A + (r0 + (lane & 15)) * K;
  floatx8 acc0 = zero8(), acc1 = zero8(), acc2 = zero8(), acc3 = zero8();
  for (int k0 = 0; k0 < K; k0 += 32) {
    __builtin_prefetch(rowp + k0 + 64, 0, 1);
    bf16x16 a  = frag_a_rowptr(rowp, k0);
    bf16x16 b0 = frag_b_global(W, Nc, k0, c0);
    bf16x16 b1 = frag_b_global(W, Nc, k0, c0 + 16);
    bf16x16 b2 = frag_b_global(W, Nc, k0, c0 + 32);
    bf16x16 b3 = frag_b_global(W, Nc, k0, c0 + 48);
    acc0 = wmma_bf16(a, b0, acc0);
    acc1 = wmma_bf16(a, b1, acc1);
    acc2 = wmma_bf16(a, b2, acc2);
    acc3 = wmma_bf16(a, b3, acc3);
  }
  store_tile<ACT, OUT_MODE>(C, Nc, r0, c0,      acc0, bias);
  store_tile<ACT, OUT_MODE>(C, Nc, r0, c0 + 16, acc1, bias);
  store_tile<ACT, OUT_MODE>(C, Nc, r0, c0 + 32, acc2, bias);
  store_tile<ACT, OUT_MODE>(C, Nc, r0, c0 + 48, acc3, bias);
}

// ---------------- front: src = emb[input_ids.T] @ We + be ----------------
__global__ void __launch_bounds__(128)
gemm_gather_bf16(const int* __restrict__ ids, const float* __restrict__ emb,
                 const float* __restrict__ W, const float* __restrict__ bias,
                 float* __restrict__ C) {
  const int grpN = D_ >> 6;
  const int totGrp = (M_ >> 4) * grpN;
  const int grp = blockIdx.x * 4 + (threadIdx.x >> 5);
  if (grp >= totGrp) return;
  const int r0 = (grp / grpN) << 4;
  const int c0 = (grp % grpN) << 6;
  const int lane = threadIdx.x & 31;
  const int row = r0 + (lane & 15);          // row = t*B + b
  const int bb = row & (B_ - 1), tt = row >> 5;
  const int id = ids[bb * T_ + tt];          // input_ids is (B,T)
  const float* rowp = emb + id * D_;
  floatx8 acc0 = zero8(), acc1 = zero8(), acc2 = zero8(), acc3 = zero8();
  for (int k0 = 0; k0 < D_; k0 += 32) {
    __builtin_prefetch(rowp + k0 + 64, 0, 1);
    bf16x16 a  = frag_a_rowptr(rowp, k0);
    bf16x16 b0 = frag_b_global(W, D_, k0, c0);
    bf16x16 b1 = frag_b_global(W, D_, k0, c0 + 16);
    bf16x16 b2 = frag_b_global(W, D_, k0, c0 + 32);
    bf16x16 b3 = frag_b_global(W, D_, k0, c0 + 48);
    acc0 = wmma_bf16(a, b0, acc0);
    acc1 = wmma_bf16(a, b1, acc1);
    acc2 = wmma_bf16(a, b2, acc2);
    acc3 = wmma_bf16(a, b3, acc3);
  }
  store_tile<0, 0>(C, D_, r0, c0,      acc0, bias);
  store_tile<0, 0>(C, D_, r0, c0 + 16, acc1, bias);
  store_tile<0, 0>(C, D_, r0, c0 + 32, acc2, bias);
  store_tile<0, 0>(C, D_, r0, c0 + 48, acc3, bias);
}

// ---------------- tiny GEMV: w3 logits = src @ Win3 + b_in3 (Nc=4) ----------------
__global__ void __launch_bounds__(256)
gemv_w3(const float* __restrict__ src, const float* __restrict__ Win3,
        const float* __restrict__ b_in3, float* __restrict__ w3l) {
  int g = blockIdx.x * blockDim.x + threadIdx.x;
  if (g >= M_ * N_) return;
  int row = g >> 2, n = g & 3;
  float acc = b_in3[n];
  for (int k = 0; k < D_; ++k) acc += src[row * D_ + k] * Win3[k * N_ + n];
  w3l[g] = acc;
}

// ---------------- recurrent scan: one workgroup per batch element ----------------
__global__ void __launch_bounds__(256)
stmae_recurrent(const float* __restrict__ v_all, const float* __restrict__ vwig_all,
                const float* __restrict__ w3l_all, const float* __restrict__ o2_all,
                float* __restrict__ o_all,
                const float* __restrict__ Wmg, const float* __restrict__ b_mg,
                const float* __restrict__ Wqkv, const float* __restrict__ b_qkv,
                const float* __restrict__ ln_g, const float* __restrict__ ln_b,
                const float* __restrict__ Wrel, const float* __restrict__ b_rel,
                const float* __restrict__ Wvb, const float* __restrict__ b_vb,
                const float* __restrict__ Wrel3, const float* __restrict__ b_rel3,
                const float* __restrict__ Mi0, const float* __restrict__ Mr0,
                const float* __restrict__ ib_p, const float* __restrict__ fb_p,
                const float* __restrict__ a1_p, const float* __restrict__ a2_p) {
  __shared__ float sMi[S_ * S_];          // 16 KB item memory
  __shared__ float sMr[N_ * S_ * R_];     // 64 KB relational memory (row-major 256x64)
  __shared__ float sGates[S_ * 2 * S_];   // 32 KB
  __shared__ float sTrans[S_ * R_];       // 16 KB
  __shared__ float sQkv[S_ * 3 * N_];     // 3 KB
  __shared__ float sQ[N_ * S_], sK[N_ * S_], sVv[N_ * S_];
  __shared__ float sV[S_], sVWig[2 * S_], sW3[N_], sA4[N_ * N_], sR[N_ * R_];
  __shared__ float sRedS[256], sRedQ[256];

  const int b = blockIdx.x;
  const int tid = threadIdx.x;
  const int wave = tid >> 5, lane = tid & 31;
  const float ib = ib_p[0], fb = fb_p[0], a1 = a1_p[0], a2 = a2_p[0];
  const int nl = lane & 15, mb = (lane >> 4) * 8, kb8 = (lane >> 4) * 8;

  for (int i = tid; i < S_ * S_; i += 256) sMi[i] = Mi0[i];
  for (int i = tid; i < N_ * S_ * R_; i += 256) sMr[i] = Mr0[i];
  __syncthreads();

  for (int t = 0; t < T_; ++t) {
    const int rb = t * B_ + b;
    if (tid < S_) sV[tid] = v_all[rb * S_ + tid];
    if (tid < 2 * S_) sVWig[tid] = vwig_all[rb * 2 * S_ + tid];
    if (tid < N_) sW3[tid] = w3l_all[rb * N_ + tid];
    __syncthreads();

    // w3 softmax (thread 0), overlapped with gates GEMM on other lanes
    if (tid == 0) {
      float mx = sW3[0];
      for (int n = 1; n < N_; ++n) mx = fmaxf(mx, sW3[n]);
      float e[N_], sum = 0.f;
      for (int n = 0; n < N_; ++n) { e[n] = expf(sW3[n] - mx); sum += e[n]; }
      for (int n = 0; n < N_; ++n) sW3[n] = e[n] / sum;
    }

    // gates = tanh(Mi) @ Wmg + b_mg + vWig  (64x128, K=64)
    // wave -> row tile (wave>>1), col group of 4 tiles ((wave&1)*64); A frag reused x4
    {
      const int r0 = (wave >> 1) * 16;
      const int cg = (wave & 1) * 64;
      floatx8 acc0 = zero8(), acc1 = zero8(), acc2 = zero8(), acc3 = zero8();
      for (int k0 = 0; k0 < S_; k0 += 32) {
        const float* p = sMi + (r0 + nl) * S_ + k0;
        bf16x16 a;
#pragma unroll
        for (int e = 0; e < 8; ++e) {
          a[e]     = (__bf16)tanhf(p[kb8 + e]);
          a[e + 8] = (__bf16)tanhf(p[16 + kb8 + e]);
        }
        bf16x16 b0 = frag_b_global(Wmg, 2 * S_, k0, cg);
        bf16x16 b1 = frag_b_global(Wmg, 2 * S_, k0, cg + 16);
        bf16x16 b2 = frag_b_global(Wmg, 2 * S_, k0, cg + 32);
        bf16x16 b3 = frag_b_global(Wmg, 2 * S_, k0, cg + 48);
        acc0 = wmma_bf16(a, b0, acc0);
        acc1 = wmma_bf16(a, b1, acc1);
        acc2 = wmma_bf16(a, b2, acc2);
        acc3 = wmma_bf16(a, b3, acc3);
      }
#pragma unroll
      for (int i = 0; i < 8; ++i) {
        int row = r0 + mb + i;
        int c0 = cg + nl;
        sGates[row * (2 * S_) + c0]      = acc0[i] + b_mg[c0]      + sVWig[c0];
        sGates[row * (2 * S_) + c0 + 16] = acc1[i] + b_mg[c0 + 16] + sVWig[c0 + 16];
        sGates[row * (2 * S_) + c0 + 32] = acc2[i] + b_mg[c0 + 32] + sVWig[c0 + 32];
        sGates[row * (2 * S_) + c0 + 48] = acc3[i] + b_mg[c0 + 48] + sVWig[c0 + 48];
      }
    }
    __syncthreads();

    // Mi = sigmoid(gf)*Mi + sigmoid(gi)*tanh(v outer v)
    for (int i = tid; i < S_ * S_; i += 256) {
      int s = i >> 6, j = i & 63;
      float gi = 1.f / (1.f + expf(-(sGates[s * 128 + j] + ib)));
      float gf = 1.f / (1.f + expf(-(sGates[s * 128 + 64 + j] + fb)));
      sMi[i] = gf * sMi[i] + gi * tanhf(sV[s] * sV[j]);
    }
    __syncthreads();

    // qkv = Mi @ Wqkv + b_qkv  (64x12, tiny -> VALU)
    for (int i = tid; i < S_ * 12; i += 256) {
      int s = i / 12, c = i % 12;
      float acc = b_qkv[c];
      for (int k = 0; k < S_; ++k) acc += sMi[s * S_ + k] * Wqkv[k * 12 + c];
      sQkv[i] = acc;
    }
    __syncthreads();

    // LayerNorm over all (S,3N)=768 entries
    {
      float ps = 0.f, pq = 0.f;
      for (int i = tid; i < S_ * 12; i += 256) { float x = sQkv[i]; ps += x; pq += x * x; }
      sRedS[tid] = ps; sRedQ[tid] = pq;
      __syncthreads();
      for (int off = 128; off > 0; off >>= 1) {
        if (tid < off) { sRedS[tid] += sRedS[tid + off]; sRedQ[tid] += sRedQ[tid + off]; }
        __syncthreads();
      }
    }
    const float mu = sRedS[0] * (1.f / 768.f);
    const float var = sRedQ[0] * (1.f / 768.f) - mu * mu;
    const float rstd = rsqrtf(var + 1e-5f);
    for (int i = tid; i < S_ * 12; i += 256) {
      int s = i / 12, c = i % 12;
      float y = (sQkv[i] - mu) * rstd * ln_g[i] + ln_b[i];
      int n = c & 3;
      if (c < 4)      sQ[n * S_ + s] = y;
      else if (c < 8) sK[n * S_ + s] = y;
      else            sVv[n * S_ + s] = y;
    }
    __syncthreads();

    // A = softmax(q k^T / sqrt(S))  (4x4)
    if (tid < 16) {
      int n = tid >> 2, m = tid & 3;
      float d = 0.f;
      for (int s = 0; s < S_; ++s) d += sQ[n * S_ + s] * sK[m * S_ + s];
      sA4[tid] = d * INV_SQRT_S;
    }
    __syncthreads();
    if (tid < 4) {
      float mx = sA4[tid * 4];
      for (int m = 1; m < 4; ++m) mx = fmaxf(mx, sA4[tid * 4 + m]);
      float e[4], sum = 0.f;
      for (int m = 0; m < 4; ++m) { e[m] = expf(sA4[tid * 4 + m] - mx); sum += e[m]; }
      for (int m = 0; m < 4; ++m) sA4[tid * 4 + m] = e[m] / sum;
    }
    __syncthreads();

    // Mr += alpha1*(Rt @ Wrel + b_rel); Rt rows synthesized on the fly:
    // Rt[(n,s), k] = sum_m A[n,m]*vv[m,s]*vv[m,k]   (256x64, K=64)
    // wave -> 2 row tiles, full 64-wide col group; synthesized A frag reused x4
    for (int rt = wave; rt < 16; rt += 8) {
      const int r0 = rt * 16;
      const int rowi = r0 + nl;
      const int nn = rowi >> 6, ss = rowi & 63;
      const float c0f = sA4[nn * 4 + 0] * sVv[0 * S_ + ss];
      const float c1f = sA4[nn * 4 + 1] * sVv[1 * S_ + ss];
      const float c2f = sA4[nn * 4 + 2] * sVv[2 * S_ + ss];
      const float c3f = sA4[nn * 4 + 3] * sVv[3 * S_ + ss];
      floatx8 acc0 = zero8(), acc1 = zero8(), acc2 = zero8(), acc3 = zero8();
      for (int k0 = 0; k0 < S_; k0 += 32) {
        bf16x16 a;
#pragma unroll
        for (int e = 0; e < 8; ++e) {
          int kA = k0 + kb8 + e, kB = k0 + 16 + kb8 + e;
          float x0 = c0f * sVv[kA] + c1f * sVv[S_ + kA] + c2f * sVv[2 * S_ + kA] + c3f * sVv[3 * S_ + kA];
          float x1 = c0f * sVv[kB] + c1f * sVv[S_ + kB] + c2f * sVv[2 * S_ + kB] + c3f * sVv[3 * S_ + kB];
          a[e] = (__bf16)x0; a[e + 8] = (__bf16)x1;
        }
        bf16x16 b0 = frag_b_global(Wrel, R_, k0, 0);
        bf16x16 b1 = frag_b_global(Wrel, R_, k0, 16);
        bf16x16 b2 = frag_b_global(Wrel, R_, k0, 32);
        bf16x16 b3 = frag_b_global(Wrel, R_, k0, 48);
        acc0 = wmma_bf16(a, b0, acc0);
        acc1 = wmma_bf16(a, b1, acc1);
        acc2 = wmma_bf16(a, b2, acc2);
        acc3 = wmma_bf16(a, b3, acc3);
      }
#pragma unroll
      for (int i = 0; i < 8; ++i) {
        int row = r0 + mb + i;                 // row = n*S + s (contiguous in sMr)
        sMr[row * R_ + nl]      += a1 * (acc0[i] + b_rel[nl]);
        sMr[row * R_ + nl + 16] += a1 * (acc1[i] + b_rel[nl + 16]);
        sMr[row * R_ + nl + 32] += a1 * (acc2[i] + b_rel[nl + 32]);
        sMr[row * R_ + nl + 48] += a1 * (acc3[i] + b_rel[nl + 48]);
      }
    }
    __syncthreads();

    // transfer = sum_n w3[n] * Mr[n]
    for (int i = tid; i < S_ * R_; i += 256) {
      float acc = 0.f;
#pragma unroll
      for (int n = 0; n < N_; ++n) acc += sW3[n] * sMr[n * S_ * R_ + i];
      sTrans[i] = acc;
    }
    __syncthreads();

    // Mi += alpha2 * tanh(transfer @ Wvb + b_vb)  (64x64, K=64)
    // wave -> row tile (wave>>1), col pair ((wave&1)*32); A frag reused x2
    {
      const int r0 = (wave >> 1) * 16;
      const int cp = (wave & 1) * 32;
      floatx8 acc0 = zero8(), acc1 = zero8();
      for (int k0 = 0; k0 < S_; k0 += 32) {
        const float* p = sTrans + (r0 + nl) * R_ + k0;
        bf16x16 a;
#pragma unroll
        for (int e = 0; e < 8; ++e) { a[e] = (__bf16)p[kb8 + e]; a[e + 8] = (__bf16)p[16 + kb8 + e]; }
        bf16x16 b0 = frag_b_global(Wvb, S_, k0, cp);
        bf16x16 b1 = frag_b_global(Wvb, S_, k0, cp + 16);
        acc0 = wmma_bf16(a, b0, acc0);
        acc1 = wmma_bf16(a, b1, acc1);
      }
#pragma unroll
      for (int i = 0; i < 8; ++i) {
        int row = r0 + mb + i;
        sMi[row * S_ + cp + nl]      += a2 * tanhf(acc0[i] + b_vb[cp + nl]);
        sMi[row * S_ + cp + nl + 16] += a2 * tanhf(acc1[i] + b_vb[cp + nl + 16]);
      }
    }
    __syncthreads();

    // r[n,j] = sum_s v[s]*Mr[n,s,j]
    {
      int n = tid >> 6, j = tid & 63;
      float acc = 0.f;
      for (int s = 0; s < S_; ++s) acc += sV[s] * sMr[n * S_ * R_ + s * R_ + j];
      sR[n * R_ + j] = acc;
    }
    __syncthreads();

    // o[j] = sum_n w3[n]*(r[n]@Wrel3 + b_rel3)[j] + o2_pre[j]
    if (tid < O_) {
      int j = tid;
      float acc = o2_all[rb * O_ + j];
      for (int n = 0; n < N_; ++n) {
        float t3 = b_rel3[j];
        for (int u = 0; u < R_; ++u) t3 += sR[n * R_ + u] * Wrel3[u * O_ + j];
        acc += sW3[n] * t3;
      }
      o_all[rb * O_ + j] = acc;
    }
    __syncthreads();
  }
}

// ---------------- final head LN (+ relu):  y = relu(LN(outs)*g + b) ----------------
__global__ void __launch_bounds__(256)
ln_relu_head(const float* __restrict__ outs, const float* __restrict__ fc_g,
             const float* __restrict__ fc_b, float* __restrict__ y) {
  __shared__ float red[256], red2[256];
  const int row = blockIdx.x;
  float ps = 0.f, pq = 0.f;
  for (int i = threadIdx.x; i < D_; i += 256) {
    float x = outs[row * D_ + i]; ps += x; pq += x * x;
  }
  red[threadIdx.x] = ps; red2[threadIdx.x] = pq;
  __syncthreads();
  for (int off = 128; off > 0; off >>= 1) {
    if (threadIdx.x < off) { red[threadIdx.x] += red[threadIdx.x + off]; red2[threadIdx.x] += red2[threadIdx.x + off]; }
    __syncthreads();
  }
  const float mu = red[0] * (1.f / D_);
  const float var = red2[0] * (1.f / D_) - mu * mu;
  const float rstd = rsqrtf(var + 1e-5f);
  for (int i = threadIdx.x; i < D_; i += 256) {
    float x = (outs[row * D_ + i] - mu) * rstd * fc_g[i] + fc_b[i];
    y[row * D_ + i] = fmaxf(x, 0.f);
  }
}

extern "C" void kernel_launch(void* const* d_in, const int* in_sizes, int n_in,
                              void* d_out, int out_size, void* d_ws, size_t ws_size,
                              hipStream_t stream) {
  (void)in_sizes; (void)n_in; (void)out_size; (void)ws_size;
  const int*   ids   = (const int*)d_in[0];
  const float* emb   = (const float*)d_in[1];
  const float* We    = (const float*)d_in[2];
  const float* be    = (const float*)d_in[3];
  const float* Win   = (const float*)d_in[4];
  const float* b_in  = (const float*)d_in[5];
  const float* Win2  = (const float*)d_in[6];
  const float* b_in2 = (const float*)d_in[7];
  const float* Win3  = (const float*)d_in[8];
  const float* b_in3 = (const float*)d_in[9];
  const float* Wig   = (const float*)d_in[10];
  const float* b_ig  = (const float*)d_in[11];
  const float* Wmg   = (const float*)d_in[12];
  const float* b_mg  = (const float*)d_in[13];
  const float* ibp   = (const float*)d_in[14];
  const float* fbp   = (const float*)d_in[15];
  const float* Wqkv  = (const float*)d_in[16];
  const float* b_qkv = (const float*)d_in[17];
  const float* ln_g  = (const float*)d_in[18];
  const float* ln_b  = (const float*)d_in[19];
  const float* a1p   = (const float*)d_in[20];
  const float* a2p   = (const float*)d_in[21];
  const float* Wrel  = (const float*)d_in[22];
  const float* b_rel = (const float*)d_in[23];
  const float* Wvb   = (const float*)d_in[24];
  const float* b_vb  = (const float*)d_in[25];
  const float* Wrel3 = (const float*)d_in[26];
  const float* b_rel3= (const float*)d_in[27];
  const float* Wm1   = (const float*)d_in[28];
  const float* b_m1  = (const float*)d_in[29];
  const float* Wm2   = (const float*)d_in[30];
  const float* b_m2  = (const float*)d_in[31];
  const float* Wout  = (const float*)d_in[32];
  const float* b_out = (const float*)d_in[33];
  const float* Mi0   = (const float*)d_in[34];
  const float* Mr0   = (const float*)d_in[35];
  const float* fc_g  = (const float*)d_in[36];
  const float* fc_b  = (const float*)d_in[37];
  const float* Wfc   = (const float*)d_in[38];
  const float* b_fc  = (const float*)d_in[39];

  float* ws   = (float*)d_ws;
  float* regA = ws;                    // 4096*512 (src -> h2 -> y)
  float* regB = regA + M_ * D_;        // 4096*512 (h1 -> outs)
  float* vwig = regB + M_ * D_;        // 4096*128
  float* vall = vwig + M_ * 2 * S_;    // 4096*64
  float* o2   = vall + M_ * S_;        // 4096*64
  float* w3l  = o2 + M_ * O_;          // 4096*4
  float* oall = w3l + M_ * N_;         // 4096*64

  // front: src = emb[ids] @ We + be
  {
    int grps = (M_ / 16) * (D_ / 64);
    gemm_gather_bf16<<<(grps + 3) / 4, 128, 0, stream>>>(ids, emb, We, be, regA);
  }
  // time-parallel projections (hoisted out of the scan)
  {
    int grps = (M_ / 16) * (S_ / 64);
    gemm_bf16_t<0, 0><<<(grps + 3) / 4, 128, 0, stream>>>(regA, Win, b_in, vall, M_, S_, D_);
    gemm_bf16_t<0, 0><<<(grps + 3) / 4, 128, 0, stream>>>(regA, Win2, b_in2, o2, M_, O_, D_);
  }
  gemv_w3<<<(M_ * N_ + 255) / 256, 256, 0, stream>>>(regA, Win3, b_in3, w3l);
  {
    int grps = (M_ / 16) * ((2 * S_) / 64);
    gemm_bf16_t<0, 0><<<(grps + 3) / 4, 128, 0, stream>>>(vall, Wig, b_ig, vwig, M_, 2 * S_, S_);
  }

  // recurrent scan: 32 workgroups (one per batch), Mi/Mr state in LDS for all T steps
  stmae_recurrent<<<B_, 256, 0, stream>>>(vall, vwig, w3l, o2, oall,
                                          Wmg, b_mg, Wqkv, b_qkv, ln_g, ln_b,
                                          Wrel, b_rel, Wvb, b_vb, Wrel3, b_rel3,
                                          Mi0, Mr0, ibp, fbp, a1p, a2p);

  // MLP tail (hoisted): h1 = relu(o@Wm1+b), h2 = relu(h1@Wm2+b), outs = h2@Wout+b
  {
    int grps = (M_ / 16) * (D_ / 64);
    gemm_bf16_t<1, 0><<<(grps + 3) / 4, 128, 0, stream>>>(oall, Wm1, b_m1, regB, M_, D_, O_);
    gemm_bf16_t<1, 0><<<(grps + 3) / 4, 128, 0, stream>>>(regB, Wm2, b_m2, regA, M_, D_, D_);
    gemm_bf16_t<0, 0><<<(grps + 3) / 4, 128, 0, stream>>>(regA, Wout, b_out, regB, M_, D_, D_);
  }
  // head: LN + relu, then logits = y @ Wfc + b_fc scattered to (B, V, T)
  ln_relu_head<<<M_, 256, 0, stream>>>(regB, fc_g, fc_b, regA);
  {
    int grps = (M_ / 16) * (V_ / 64);
    gemm_bf16_t<0, 1><<<(grps + 3) / 4, 128, 0, stream>>>(regA, Wfc, b_fc, (float*)d_out,
                                                          M_, V_, D_);
  }
}
